// MultiHeadAttention_77060303225024
// MI455X (gfx1250) — matrix operand order repaired and checked by reading, not compiled, not run on hardware
//
#include <hip/hip_runtime.h>
#include <stdint.h>

#define B_SZ   2
#define L_SZ   2048
#define DMODEL 2048
#define NH     16
#define HD     128

typedef __attribute__((ext_vector_type(16))) __bf16 v16bf;
typedef __attribute__((ext_vector_type(8)))  float  v8f;
typedef unsigned int u32x4 __attribute__((ext_vector_type(4)));
typedef int          i32x4 __attribute__((ext_vector_type(4)));
typedef int          i32x8 __attribute__((ext_vector_type(8)));

union BF16x16 { v16bf v; uint4 q[2]; };

__device__ __forceinline__ __bf16 f2bf(float f) {
  union { float f; uint32_t u; } x; x.f = f;
  uint32_t r = (x.u + 0x7FFFu + ((x.u >> 16) & 1u)) >> 16;
  union { uint16_t s; __bf16 b; } y; y.s = (uint16_t)r;
  return y.b;
}

// A-operand (16x32 bf16, MxK): lane half 0 -> K k..k+15,
// half 1 -> K k+16..k+31; row = lane&15 (caller indexes row ptr).
__device__ __forceinline__ v16bf loadA16(const __bf16* row, int k, int half) {
  BF16x16 r;
  const __bf16* p = row + k + half * 16;
  r.q[0] = *(const uint4*)(p);
  r.q[1] = *(const uint4*)(p + 8);
  return r.v;
}

// B-operand (32x16 bf16, KxN): lane = column; half 0 -> K k..k+15,
// half 1 -> K k+16..k+31, contiguous in memory (row-major W / V^T rows).
__device__ __forceinline__ v16bf loadB16(const __bf16* row, int k, int half) {
  BF16x16 r;
  const uint4* p = (const uint4*)(row + k + half * 16);
  r.q[0] = p[0];
  r.q[1] = p[1];
  return r.v;
}

__device__ __forceinline__ v8f wmma_bf16(v16bf a, v16bf b, v8f c) {
  return __builtin_amdgcn_wmma_f32_16x16x32_bf16(false, a, false, b, (short)0, c,
                                                 false, false);
}

#if __has_builtin(__builtin_amdgcn_tensor_load_to_lds)
#define USE_TDM 1
#else
#define USE_TDM 0
#endif

template <int N>
__device__ __forceinline__ void wait_tensorcnt() {
#if __has_builtin(__builtin_amdgcn_s_wait_tensorcnt)
  __builtin_amdgcn_s_wait_tensorcnt((short)N);
#else
  asm volatile("s_wait_tensorcnt %0" ::"i"(N));
#endif
}

#if USE_TDM
// Issue a TDM 2-D tile load: tile_rows x tile_k bf16 elements from a row-major
// tensor (row stride = row_stride elems) into contiguous LDS at lds_addr.
// D# layout per CDNA5 ISA ch.8 (async_tensor): group0 = control/addresses,
// group1 = dims/strides, groups 2-3 zero (<=2D tile).
__device__ __forceinline__ void tdm_load_2d(uint32_t lds_addr, const void* gptr,
                                            uint32_t tile_k, uint32_t tile_rows,
                                            uint32_t row_stride,
                                            uint32_t tensor_d0,
                                            uint32_t tensor_d1) {
  uint64_t ga = (uint64_t)(uintptr_t)gptr;
  u32x4 g0;
  g0[0] = 1u;                                               // count=1, user D#
  g0[1] = lds_addr;                                         // LDS byte address
  g0[2] = (uint32_t)ga;                                     // global_addr[31:0]
  g0[3] = (uint32_t)((ga >> 32) & 0x01FFFFFFu) | (2u << 30);// [56:32] | type=2
  i32x8 g1;
  g1[0] = (int)(1u << 16);                                  // data_size=1 (2B)
  g1[1] = (int)((tensor_d0 & 0xFFFFu) << 16);               // tensor_dim0 lo16
  g1[2] = (int)((tensor_d0 >> 16) | ((tensor_d1 & 0xFFFFu) << 16));
  g1[3] = (int)((tensor_d1 >> 16) | (tile_k << 16));        // tile_dim0
  g1[4] = (int)(tile_rows & 0xFFFFu);                       // tile_dim1
  g1[5] = (int)row_stride;                                  // dim0_stride[31:0]
  g1[6] = 0;
  g1[7] = 0;
  i32x4 gz = {0, 0, 0, 0};
#if __clang_major__ >= 23
  i32x8 gz8 = {0, 0, 0, 0, 0, 0, 0, 0};
  __builtin_amdgcn_tensor_load_to_lds(g0, g1, gz, gz, gz8, 0);
#else
  __builtin_amdgcn_tensor_load_to_lds(g0, g1, gz, gz, 0);
#endif
}
#endif  // USE_TDM

__global__ void cvt_f32_bf16(const float* __restrict__ src,
                             __bf16* __restrict__ dst, int n) {
  int i = blockIdx.x * blockDim.x + threadIdx.x;
  int stride = gridDim.x * blockDim.x;
  for (; i < n; i += stride) dst[i] = f2bf(src[i]);
}

// MODE 0: q/k projection + fused RoPE -> bf16 [b,h,l,d]
// MODE 1: v projection -> bf16 transposed [b,h,d,l]
// MODE 2: output projection -> fp32 [m,n]
// W tiles (64 rows x 32 K) are staged into LDS by the Tensor Data Mover,
// double-buffered per wave, completion tracked with TENSORcnt.
template <int MODE>
__global__ __launch_bounds__(128) void gemm_bf16_wmma(
    const __bf16* __restrict__ A, const __bf16* __restrict__ W,
    void* __restrict__ dst, int M, int N, int K) {
#if USE_TDM
  __shared__ __align__(16) __bf16 stage[4][2][64 * 32];
#endif
  const int lane = threadIdx.x & 31;
  const int wave = threadIdx.x >> 5;
  const int half = lane >> 4;
  const int ln   = lane & 15;
  const int m0 = (blockIdx.y * 4 + wave) * 16;
  const int n0 = blockIdx.x * 64;
  if (m0 >= M) return;

  v8f acc[4];
#pragma unroll
  for (int t = 0; t < 4; ++t) acc[t] = (v8f){0, 0, 0, 0, 0, 0, 0, 0};

  const __bf16* arow = A + (size_t)(m0 + ln) * K;

#if USE_TDM
  const __bf16* wtile = W + (size_t)n0 * K;  // rows n0..n0+63
  uint32_t ldsb[2];
  ldsb[0] = (uint32_t)(uintptr_t)&stage[wave][0][0];
  ldsb[1] = (uint32_t)(uintptr_t)&stage[wave][1][0];
  tdm_load_2d(ldsb[0], wtile, 32, 64, (uint32_t)K, (uint32_t)K, (uint32_t)N);
  for (int k = 0; k < K; k += 32) {
    int cur = (k >> 5) & 1;
    if (k + 32 < K) {
      tdm_load_2d(ldsb[cur ^ 1], wtile + (k + 32), 32, 64, (uint32_t)K,
                  (uint32_t)K, (uint32_t)N);
      wait_tensorcnt<1>();  // buffer `cur` complete (TDM completes in order)
    } else {
      wait_tensorcnt<0>();
    }
    v16bf a = loadA16(arow, k, half);
    const __bf16* sb = &stage[wave][cur][0];
#pragma unroll
    for (int t = 0; t < 4; ++t) {
      v16bf b = loadB16(sb + (t * 16 + ln) * 32, 0, half);
      acc[t] = wmma_bf16(a, b, acc[t]);
    }
  }
#else
  for (int k = 0; k < K; k += 32) {
    v16bf a = loadA16(arow, k, half);
#pragma unroll
    for (int t = 0; t < 4; ++t) {
      const __bf16* wrow = W + (size_t)(n0 + t * 16 + ln) * K;
      v16bf b = loadB16(wrow, k, half);
      acc[t] = wmma_bf16(a, b, acc[t]);
    }
  }
#endif

#pragma unroll
  for (int t = 0; t < 4; ++t) {
    int n = n0 + t * 16 + ln;  // lane parity == column parity
#pragma unroll
    for (int r = 0; r < 8; ++r) {
      int m = m0 + r + half * 8;
      float v = acc[t][r];
      if (MODE == 2) {
        ((float*)dst)[(size_t)m * N + n] = v;
      } else {
        int b = m / L_SZ, l = m % L_SZ;
        int h = n / HD, d = n % HD;
        if (MODE == 0) {
          float part = __shfl_xor(v, 1, 32);  // RoPE partner lives in lane^1
          int p = d >> 1;
          float fr = __powf(10000.0f, -(float)(2 * p) / (float)HD);
          float ang = (float)l * fr, sn, cs;
          __sincosf(ang, &sn, &cs);
          float o = (d & 1) ? (part * sn + v * cs) : (v * cs - part * sn);
          ((__bf16*)dst)[(((size_t)b * NH + h) * L_SZ + l) * HD + d] = f2bf(o);
        } else {  // MODE 1: V^T per head
          ((__bf16*)dst)[(((size_t)b * NH + h) * HD + d) * L_SZ + l] = f2bf(v);
        }
      }
    }
  }
}

// One workgroup = 16 query rows of one (b,h). Dynamic LDS:
//   sS  : float [16][2048]  scores / probabilities      (128 KB)
//   sP  : bf16  [16][2048]  normalized probs for P*V    ( 64 KB)
//   sred: float [16][8], rstat: float[32]
__global__ __launch_bounds__(128) void attn_wmma(
    const __bf16* __restrict__ qb, const __bf16* __restrict__ kb,
    const __bf16* __restrict__ vT, float* __restrict__ attn_out,
    __bf16* __restrict__ o_bf) {
  extern __shared__ char smem[];
  float*  sS   = (float*)smem;
  __bf16* sP   = (__bf16*)(smem + 16 * 2048 * 4);
  float*  sred = (float*)(smem + 16 * 2048 * 4 + 16 * 2048 * 2);
  float*  rstat = sred + 16 * 8;

  const int t    = threadIdx.x;
  const int lane = t & 31, wave = t >> 5;
  const int half = lane >> 4, ln = lane & 15;
  const int m0    = blockIdx.x * 16;
  const int bh    = blockIdx.y;  // b*NH + h
  const int ncols = m0 + 16;     // causal: only columns <= last row
  const int ntiles = ncols >> 4;

  const __bf16* qbase = qb + (size_t)bh * L_SZ * HD;
  const __bf16* kbase = kb + (size_t)bh * L_SZ * HD;
  const __bf16* vbase = vT + (size_t)bh * HD * L_SZ;

  // q A-operands for the whole stripe (HD=128 => 4 K-chunks of 32)
  v16bf qa[4];
  {
    const __bf16* qrow = qbase + (size_t)(m0 + ln) * HD;
#pragma unroll
    for (int kk = 0; kk < 4; ++kk) qa[kk] = loadA16(qrow, kk * 32, half);
  }

  // Phase 1: scores = scale * q k^T + causal mask -> LDS
  const float scale = 0.08838834764831845f;  // 1/sqrt(128)
  for (int tc = wave; tc < ntiles; tc += 4) {
    v8f acc = (v8f){0, 0, 0, 0, 0, 0, 0, 0};
    const __bf16* krow = kbase + (size_t)(tc * 16 + ln) * HD;
#pragma unroll
    for (int kk = 0; kk < 4; ++kk)
      acc = wmma_bf16(qa[kk], loadB16(krow, kk * 32, half), acc);
#pragma unroll
    for (int r = 0; r < 8; ++r) {
      int row = r + half * 8;
      int c = tc * 16 + ln;
      float val = acc[r] * scale + ((c <= m0 + row) ? 0.0f : -1.0e9f);
      sS[row * 2048 + c] = val;
    }
  }
  __syncthreads();

  const int row = t & 15, seg = t >> 4;
  // Phase 2a: row max
  {
    float mx = -3.4e38f;
    for (int c = seg; c < ncols; c += 8) mx = fmaxf(mx, sS[row * 2048 + c]);
    sred[row * 8 + seg] = mx;
  }
  __syncthreads();
  if (t < 16) {
    float mx = sred[t * 8];
    for (int j = 1; j < 8; ++j) mx = fmaxf(mx, sred[t * 8 + j]);
    rstat[t] = mx;
  }
  __syncthreads();
  // Phase 2b: exp + row sum (in place)
  {
    float mx = rstat[row], s = 0.f;
    for (int c = seg; c < ncols; c += 8) {
      float p = __expf(sS[row * 2048 + c] - mx);
      sS[row * 2048 + c] = p;
      s += p;
    }
    sred[row * 8 + seg] = s;
  }
  __syncthreads();
  if (t < 16) {
    float s = 0.f;
    for (int j = 0; j < 8; ++j) s += sred[t * 8 + j];
    rstat[16 + t] = s;
  }
  __syncthreads();
  // Phase 3: write attn (zeros above diagonal) + pack bf16 probs
  {
    float inv = 1.0f / rstat[16 + row];
    size_t abase = ((size_t)bh * L_SZ + (m0 + row)) * L_SZ;
    for (int c = seg; c < ncols; c += 8) {
      float a = sS[row * 2048 + c] * inv;
      attn_out[abase + c] = a;
      sP[row * 2048 + c] = f2bf(a);
    }
    for (int c = ncols + seg; c < L_SZ; c += 8) attn_out[abase + c] = 0.0f;
  }
  // zero-pad sP to a multiple of K=32
  {
    int sround = (ncols + 31) & ~31;
    for (int idx = t; idx < 16 * (sround - ncols); idx += 128) {
      int rr = idx >> 4;
      int cc = ncols + (idx & 15);
      sP[rr * 2048 + cc] = f2bf(0.0f);
    }
  }
  __syncthreads();

  // Phase 4: out = P @ V  (A from LDS, B = V^T rows from global)
  int sround = (ncols + 31) & ~31;
  int bb = bh / NH, hh = bh % NH;
  for (int dt = wave; dt < 8; dt += 4) {
    int d0 = dt * 16;
    v8f acc = (v8f){0, 0, 0, 0, 0, 0, 0, 0};
    const __bf16* vrow = vbase + (size_t)(d0 + ln) * L_SZ;
    for (int s = 0; s < sround; s += 32) {
      v16bf a = loadA16(sP + ln * 2048, s, half);
      v16bf b = loadB16(vrow, s, half);
      acc = wmma_bf16(a, b, acc);
    }
#pragma unroll
    for (int r = 0; r < 8; ++r) {
      int l = m0 + r + half * 8;
      o_bf[((size_t)bb * L_SZ + l) * DMODEL + hh * HD + d0 + ln] = f2bf(acc[r]);
    }
  }
}

extern "C" void kernel_launch(void* const* d_in, const int* in_sizes, int n_in,
                              void* d_out, int out_size, void* d_ws,
                              size_t ws_size, hipStream_t stream) {
  (void)in_sizes; (void)n_in; (void)out_size; (void)ws_size;
  const float* x_q  = (const float*)d_in[0];
  const float* x_kv = (const float*)d_in[1];
  // d_in[2] = attn_mask: causal -1e9 mask reproduced analytically
  const float* w_q = (const float*)d_in[3];
  const float* w_k = (const float*)d_in[4];
  const float* w_v = (const float*)d_in[5];
  const float* w_o = (const float*)d_in[6];

  const size_t nBLD = (size_t)B_SZ * L_SZ * DMODEL;  // 8,388,608
  const size_t nDD  = (size_t)DMODEL * DMODEL;       // 4,194,304

  char* ws = (char*)d_ws;
  __bf16* xq_bf  = (__bf16*)ws; ws += nBLD * 2;
  __bf16* xkv_bf = (__bf16*)ws; ws += nBLD * 2;
  __bf16* wq_bf  = (__bf16*)ws; ws += nDD * 2;
  __bf16* wk_bf  = (__bf16*)ws; ws += nDD * 2;
  __bf16* wv_bf  = (__bf16*)ws; ws += nDD * 2;
  __bf16* wo_bf  = (__bf16*)ws; ws += nDD * 2;
  __bf16* q_bf   = (__bf16*)ws; ws += nBLD * 2;  // [b,h,l,d] post-RoPE
  __bf16* k_bf   = (__bf16*)ws; ws += nBLD * 2;  // [b,h,l,d] post-RoPE
  __bf16* vT_bf  = (__bf16*)ws; ws += nBLD * 2;  // [b,h,d,l]
  __bf16* o_bf   = (__bf16*)ws; ws += nBLD * 2;  // [b,l,h*d]

  float* out = (float*)d_out;
  float* attn_out = out + nBLD;

  cvt_f32_bf16<<<4096, 256, 0, stream>>>(x_q, xq_bf, (int)nBLD);
  cvt_f32_bf16<<<4096, 256, 0, stream>>>(x_kv, xkv_bf, (int)nBLD);
  cvt_f32_bf16<<<2048, 256, 0, stream>>>(w_q, wq_bf, (int)nDD);
  cvt_f32_bf16<<<2048, 256, 0, stream>>>(w_k, wk_bf, (int)nDD);
  cvt_f32_bf16<<<2048, 256, 0, stream>>>(w_v, wv_bf, (int)nDD);
  cvt_f32_bf16<<<2048, 256, 0, stream>>>(w_o, wo_bf, (int)nDD);

  dim3 gblock(128);
  dim3 ggrid(DMODEL / 64, (B_SZ * L_SZ) / 64);
  gemm_bf16_wmma<0><<<ggrid, gblock, 0, stream>>>(xq_bf, wq_bf, q_bf,
                                                  B_SZ * L_SZ, DMODEL, DMODEL);
  gemm_bf16_wmma<0><<<ggrid, gblock, 0, stream>>>(xkv_bf, wk_bf, k_bf,
                                                  B_SZ * L_SZ, DMODEL, DMODEL);
  gemm_bf16_wmma<1><<<ggrid, gblock, 0, stream>>>(xkv_bf, wv_bf, vT_bf,
                                                  B_SZ * L_SZ, DMODEL, DMODEL);

  size_t smem = (size_t)16 * 2048 * 4 + 16 * 2048 * 2 + (16 * 8 + 32) * 4;
  attn_wmma<<<dim3(L_SZ / 16, B_SZ * NH), 128, smem, stream>>>(
      q_bf, k_bf, vT_bf, attn_out, o_bf);

  gemm_bf16_wmma<2><<<ggrid, gblock, 0, stream>>>(o_bf, wo_bf, (void*)out,
                                                  B_SZ * L_SZ, DMODEL, DMODEL);
}
